// Diffusion_35691178229986
// MI455X (gfx1250) — matrix operand order repaired and checked by reading, not compiled, not run on hardware
//
#include <hip/hip_runtime.h>
#include <hip/hip_bf16.h>

typedef __attribute__((ext_vector_type(16))) _Float16 v16h;
typedef __attribute__((ext_vector_type(8)))  float    v8f;
typedef __attribute__((ext_vector_type(4)))  unsigned tdm_u32x4;
typedef __attribute__((ext_vector_type(8)))  int      tdm_i32x8;
typedef __attribute__((ext_vector_type(4)))  int      tdm_i32x4;

union Frag {
    v16h     v;
    uint4    q[2];
    unsigned u[8];
    _Float16 h[16];
};

__device__ __forceinline__ v8f wmma_f16(v16h a, v16h b, v8f c) {
    return __builtin_amdgcn_wmma_f32_16x16x32_f16(
        false, a, false, b, (short)0, c, false, false);
}

// mish(x) = x * tanh(softplus(x)) ; tanh(log(1+e)) = e(e+2)/(e(e+2)+2)
__device__ __forceinline__ float mishf(float x) {
    float e = __expf(x);
    float n = e * (e + 2.f);
    float t = n * __builtin_amdgcn_rcpf(n + 2.f);
    return (x > 20.f) ? x : x * t;
}

__device__ __forceinline__ unsigned pcg(unsigned v) {
    v = v * 747796405u + 2891336453u;
    unsigned w = ((v >> ((v >> 28u) + 4u)) ^ v) * 277803737u;
    return (w >> 22u) ^ w;
}

__device__ __forceinline__ float rngn(unsigned idx, unsigned salt) {
    unsigned a = pcg(idx ^ (salt * 0x9E3779B9u));
    unsigned b = pcg(a + 0x85EBCA6Bu);
    float u1 = fmaxf((a >> 8) * (1.f / 16777216.f), 1e-7f);
    float u2 = (b >> 8) * (1.f / 16777216.f);
    return sqrtf(-2.f * __logf(u1)) * __cosf(6.28318530718f * u2);
}

// ---------------- constants ----------------
#define BATCH 32768
#define SDIM  256
#define ADIM  32
#define HDIM  256
#define TSTEPS 20

// half offsets inside f16 weight arena: per tile = [kt][nt][lane(32)][16]
#define OFF_W1h 0          // 9 ktiles x 16 ntiles x 512
#define OFF_W2h 73728      // 8 x 16 x 512
#define OFF_W3h 139264
#define OFF_WOh 204800     // 8 x 2 x 512
#define WFH_TOTAL 212992

// ---------------- prep: weights -> f16 B-fragment swizzle ----------------
// lane (g,ln) holds column N0+ln, j=0..7 -> K=K0+8g+j, j=8..15 -> K=K0+8g+16+(j-8)
extern "C" __global__ void diff_prep_w(const float* __restrict__ W,
                                       _Float16* __restrict__ dst,
                                       int ktiles, int ntiles, int ldw) {
    int idx = blockIdx.x * 256 + threadIdx.x;
    int total = ktiles * ntiles * 512;
    if (idx >= total) return;
    int j    = idx & 15;
    int lane = (idx >> 4) & 31;
    int t    = idx >> 9;
    int nt   = t % ntiles;
    int kt   = t / ntiles;
    int gg = lane >> 4, ll = lane & 15;
    int k = kt * 32 + 8 * gg + ((j < 8) ? j : 16 + (j - 8));
    int n = nt * 16 + ll;
    dst[idx] = (_Float16)W[k * ldw + n];
}

// ---------------- prep: state -> f16 A-fragment swizzle ----------------
// ISA A layout: lane (g,ln) = row ln; j=0..7 -> K=8g+j, j=8..15 -> K=8g+16+(j-8)
extern "C" __global__ void diff_prep_state(const float* __restrict__ S,
                                           _Float16* __restrict__ dst) {
    int idx = blockIdx.x * 256 + threadIdx.x;
    int j    = idx & 15;
    int lane = (idx >> 4) & 31;
    int kt   = (idx >> 9) & 7;
    int rt   = idx >> 12;
    int gg = lane >> 4, ll = lane & 15;
    int koff = kt * 32 + 8 * gg + ((j < 8) ? j : 16 + (j - 8));
    int row  = rt * 16 + ll;
    dst[idx] = (_Float16)S[row * SDIM + koff];
}

// ---------------- diffusion schedule ----------------
extern "C" __global__ void diff_sched(float* __restrict__ sched) {
    if (threadIdx.x == 0 && blockIdx.x == 0) {
        float ac = 1.f;
        for (int i = 0; i < TSTEPS; ++i) {
            float beta  = 1e-4f + (0.02f - 1e-4f) * (float)i / 19.f;
            float alpha = 1.f - beta;
            float ac_prev = ac;
            ac *= alpha;
            sched[i]      = sqrtf(1.f / ac);
            sched[20 + i] = sqrtf(1.f / ac - 1.f);
            float plv = __logf(fmaxf(beta * (1.f - ac_prev) / (1.f - ac), 1e-20f));
            sched[40 + i] = __expf(0.5f * plv);
            sched[60 + i] = beta * sqrtf(ac_prev) / (1.f - ac);
            sched[80 + i] = (1.f - ac_prev) * sqrtf(alpha) / (1.f - ac);
        }
    }
}

// ---------------- time-embedding folded into layer-1 bias ----------------
extern "C" __global__ void diff_temb(const float* __restrict__ Wt1, const float* __restrict__ bt1,
                                     const float* __restrict__ Wt2, const float* __restrict__ bt2,
                                     const float* __restrict__ W1,  const float* __restrict__ b1,
                                     float* __restrict__ effb1) {
    int s = blockIdx.x;
    int n = threadIdx.x;
    float te0[16];
    #pragma unroll
    for (int i = 0; i < 8; ++i) {
        float freq = __expf((float)i * (-9.21034037198f / 7.f));
        float ang = (float)s * freq;
        te0[i]     = __sinf(ang);
        te0[8 + i] = __cosf(ang);
    }
    float h1[32];
    #pragma unroll
    for (int jj = 0; jj < 32; ++jj) {
        float v = bt1[jj];
        #pragma unroll
        for (int i = 0; i < 16; ++i) v += te0[i] * Wt1[i * 32 + jj];
        h1[jj] = mishf(v);
    }
    float acc = b1[n];
    #pragma unroll
    for (int k = 0; k < 16; ++k) {
        float te = bt2[k];
        #pragma unroll
        for (int i = 0; i < 32; ++i) te += h1[i] * Wt2[i * 16 + k];
        acc += te * W1[(SDIM + ADIM + k) * HDIM + n];
    }
    effb1[s * 256 + n] = acc;
}

// ---------------- x_T init ----------------
extern "C" __global__ void diff_init_x(float* __restrict__ x) {
    int i = blockIdx.x * 256 + threadIdx.x;
    x[i] = rngn((unsigned)i, 777u);
}

// ---------------- step-kernel helpers ----------------
__device__ __forceinline__ void scatter_mish(_Float16* hs, v8f acc, int nt, int g, int ln) {
    int N = nt * 16 + ln, kt2 = N >> 5, koff = N & 31;
    int g2 = (koff >> 3) & 1;
    int j  = (koff & 7) + 8 * (koff >> 4);
    #pragma unroll
    for (int r = 0; r < 8; ++r)
        hs[((kt2 * 32 + g2 * 16 + (r + 8 * g)) * 16) + j] = (_Float16)mishf(acc[r]);
}

__device__ __forceinline__ void load_act(Frag (&a)[8], const _Float16* hs, int lane) {
    #pragma unroll
    for (int kt = 0; kt < 8; ++kt)
        a[kt].v = *(const v16h*)(hs + (kt * 32 + lane) * 16);
}

// 256->256 layer for two 16-row slabs; WOFFH is a compile-time half-offset so
// all weight loads fold to  global_load_b128 v, v_laneoff, s[wf] offset:IMM
template<int WOFFH>
__device__ __forceinline__ void layer_hh(const _Float16* __restrict__ wlane,
                                         const float* __restrict__ bias,
                                         _Float16* hs0, _Float16* hs1,
                                         const Frag (&a0)[8], const Frag (&a1)[8],
                                         int g, int ln) {
    #pragma unroll
    for (int nt = 0; nt < 16; ++nt) {
        float bv = bias[nt * 16 + ln];
        v8f acc0 = {bv, bv, bv, bv, bv, bv, bv, bv};
        v8f acc1 = acc0;
        #pragma unroll
        for (int kt = 0; kt < 8; ++kt) {
            Frag b;
            b.v = *(const v16h*)(wlane + WOFFH + (kt * 16 + nt) * 512);
            acc0 = wmma_f16(a0[kt].v, b.v, acc0);   // two independent chains per
            acc1 = wmma_f16(a1[kt].v, b.v, acc1);   // B-frag: fills hazard slots
        }
        scatter_mish(hs0, acc0, nt, g, ln);
        scatter_mish(hs1, acc1, nt, g, ln);
    }
}

// ---------------- main per-step kernel ----------------
// 64 threads = 2 waves; each wave owns two 16-row slabs (32 rows); 64 rows/block.
extern "C" __global__ void __launch_bounds__(64)
diff_step(const _Float16* __restrict__ wf, const _Float16* __restrict__ sf16,
          const float* __restrict__ effb1, const float* __restrict__ b2,
          const float* __restrict__ b3,    const float* __restrict__ bo,
          const float* __restrict__ sched, float* __restrict__ x,
          float* __restrict__ out, int s) {
    __shared__ _Float16 sh_h[4][8 * 32 * 16];   // 4 slabs x 8 KB
    __shared__ float    sh_x[64 * ADIM];        // 8 KB

    const int tid  = threadIdx.x;
    const int wave = tid >> 5;
    const int lane = tid & 31;
    const int g    = lane >> 4;
    const int ln   = lane & 15;
    const int row0 = blockIdx.x * 64;

#ifdef __AMDGCN__
    // ---- TDM: async DMA of this block's x tile (2048 contiguous f32) into LDS ----
    if (wave == 0) {
        unsigned ldsa = (unsigned)(size_t)(void*)&sh_x[0];
        unsigned long long ga = (unsigned long long)(size_t)(x + (size_t)row0 * ADIM);
        tdm_u32x4 g0 = { 1u,                                   // count=1, flags=0
                         ldsa,                                 // lds_addr
                         (unsigned)ga,                         // global_addr lo
                         (unsigned)(ga >> 32) | (2u << 30) };  // ga hi | type=2
        tdm_i32x8 g1 = { (int)(2u << 16),      // data_size = 4B
                         (int)(2048u << 16),   // tensor_dim0 = 2048
                         (int)(1u << 16),      // tensor_dim1 = 1
                         (int)(2048u << 16),   // tile_dim0 = 2048
                         1,                    // tile_dim1 = 1
                         2048,                 // tensor_dim0_stride
                         0, 0 };
        tdm_i32x4 z4 = {0, 0, 0, 0};
#if __clang_major__ >= 23
        tdm_i32x8 z8 = {0, 0, 0, 0, 0, 0, 0, 0};
        __builtin_amdgcn_tensor_load_to_lds(g0, g1, z4, z4, z8, 0);
#else
        __builtin_amdgcn_tensor_load_to_lds(g0, g1, z4, z4, 0);
#endif
        __builtin_amdgcn_s_wait_tensorcnt(0);
    }
#endif
    __syncthreads();

    const float srp = sched[s],      srm = sched[20 + s], sig = sched[40 + s];
    const float c1  = sched[60 + s], c2  = sched[80 + s];
    const float* eb1 = effb1 + s * 256;

    const _Float16* wlane = wf + (size_t)lane * 16;   // single VGPR byte offset
    const int lrow0 = wave * 32;                      // slab0 local row base
    _Float16* hs0 = &sh_h[wave * 2 + 0][0];
    _Float16* hs1 = &sh_h[wave * 2 + 1][0];

    // x A-fragments (K-tile of 32) for both slabs, from f32 in LDS
    Frag ax0, ax1;
    {
        const float* xr0 = sh_x + (lrow0 + ln) * ADIM;
        const float* xr1 = xr0 + 16 * ADIM;
        #pragma unroll
        for (int j = 0; j < 16; j += 2) {
            int koff = 8 * g + ((j < 8) ? j : 16 + (j - 8));
            ax0.h[j]     = (_Float16)xr0[koff];
            ax0.h[j + 1] = (_Float16)xr0[koff + 1];
            ax1.h[j]     = (_Float16)xr1[koff];
            ax1.h[j + 1] = (_Float16)xr1[koff + 1];
        }
    }

    // state A-fragments, pre-swizzled: one 32B coalesced load per lane per frag
    const size_t grt0 = (size_t)((row0 + lrow0) >> 4);
    Frag a0[8], a1[8];
    #pragma unroll
    for (int kt = 0; kt < 8; ++kt) {
        a0[kt].v = *(const v16h*)(sf16 + ((grt0 * 8 + kt) * 32 + lane) * 16);
        a1[kt].v = *(const v16h*)(sf16 + (((grt0 + 1) * 8 + kt) * 32 + lane) * 16);
    }

    // ---- layer 1: [x|state] @ W1 + eff_b1, mish ----
    #pragma unroll
    for (int nt = 0; nt < 16; ++nt) {
        float bv = eb1[nt * 16 + ln];
        v8f acc0 = {bv, bv, bv, bv, bv, bv, bv, bv};
        v8f acc1 = acc0;
        Frag b;
        b.v = *(const v16h*)(wlane + OFF_W1h + nt * 512);           // x K-tile
        acc0 = wmma_f16(ax0.v, b.v, acc0);
        acc1 = wmma_f16(ax1.v, b.v, acc1);
        #pragma unroll
        for (int kt = 0; kt < 8; ++kt) {
            b.v = *(const v16h*)(wlane + OFF_W1h + ((1 + kt) * 16 + nt) * 512);
            acc0 = wmma_f16(a0[kt].v, b.v, acc0);
            acc1 = wmma_f16(a1[kt].v, b.v, acc1);
        }
        scatter_mish(hs0, acc0, nt, g, ln);
        scatter_mish(hs1, acc1, nt, g, ln);
    }

    // ---- layers 2, 3 (compile-time weight offsets) ----
    load_act(a0, hs0, lane);
    load_act(a1, hs1, lane);
    layer_hh<OFF_W2h>(wlane, b2, hs0, hs1, a0, a1, g, ln);
    load_act(a0, hs0, lane);
    load_act(a1, hs1, lane);
    layer_hh<OFF_W3h>(wlane, b3, hs0, hs1, a0, a1, g, ln);
    load_act(a0, hs0, lane);
    load_act(a1, hs1, lane);

    // ---- output layer + diffusion update, both slabs ----
    #pragma unroll
    for (int p = 0; p < 2; ++p) {
        const int lrow = lrow0 + p * 16;
        #pragma unroll
        for (int nt = 0; nt < 2; ++nt) {
            float bv = bo[nt * 16 + ln];
            v8f acc = {bv, bv, bv, bv, bv, bv, bv, bv};
            #pragma unroll
            for (int kt = 0; kt < 8; ++kt) {
                Frag b;
                b.v = *(const v16h*)(wlane + OFF_WOh + (kt * 2 + nt) * 512);
                acc = wmma_f16((p ? a1 : a0)[kt].v, b.v, acc);
            }
            #pragma unroll
            for (int r = 0; r < 8; ++r) {
                int M   = lrow + r + 8 * g;
                int col = nt * 16 + ln;
                float nh = acc[r];
                float xv = sh_x[M * ADIM + col];
                float xrec = fminf(fmaxf(srp * xv - srm * nh, -1.f), 1.f);
                float mean = c1 * xv + c2 * xrec;
                int gidx = (row0 + M) * ADIM + col;
                if (s != 0) {
                    float nz = rngn((unsigned)gidx, (unsigned)(1000 + s));
                    x[gidx] = mean + sig * nz;      // row-wise dependence: in-place safe
                } else {
                    out[gidx] = fminf(fmaxf(mean, -1.f), 1.f);
                }
            }
        }
    }
}

// ---------------- launch ----------------
extern "C" void kernel_launch(void* const* d_in, const int* in_sizes, int n_in,
                              void* d_out, int out_size, void* d_ws, size_t ws_size,
                              hipStream_t stream) {
    const float* state = (const float*)d_in[0];
    const float* Wt1 = (const float*)d_in[1];
    const float* bt1 = (const float*)d_in[2];
    const float* Wt2 = (const float*)d_in[3];
    const float* bt2 = (const float*)d_in[4];
    const float* W1  = (const float*)d_in[5];
    const float* b1  = (const float*)d_in[6];
    const float* W2  = (const float*)d_in[7];
    const float* b2  = (const float*)d_in[8];
    const float* W3  = (const float*)d_in[9];
    const float* b3  = (const float*)d_in[10];
    const float* Wo  = (const float*)d_in[11];
    const float* bo  = (const float*)d_in[12];
    float* out = (float*)d_out;

    char* ws = (char*)d_ws;
    const size_t OFF_WF    = 0;
    const size_t OFF_SF16  = OFF_WF + (size_t)WFH_TOTAL * 2;
    const size_t OFF_EFFB  = OFF_SF16 + (size_t)BATCH * SDIM * 2;
    const size_t OFF_SCHED = OFF_EFFB + (size_t)TSTEPS * 256 * 4;
    const size_t OFF_X     = OFF_SCHED + 512;

    _Float16* wf    = (_Float16*)(ws + OFF_WF);
    _Float16* sf16  = (_Float16*)(ws + OFF_SF16);
    float*    effb1 = (float*)(ws + OFF_EFFB);
    float*    sched = (float*)(ws + OFF_SCHED);
    float*    x     = (float*)(ws + OFF_X);

    diff_prep_w<<<(9 * 16 * 512) / 256, 256, 0, stream>>>(W1, wf + OFF_W1h, 9, 16, HDIM);
    diff_prep_w<<<(8 * 16 * 512) / 256, 256, 0, stream>>>(W2, wf + OFF_W2h, 8, 16, HDIM);
    diff_prep_w<<<(8 * 16 * 512) / 256, 256, 0, stream>>>(W3, wf + OFF_W3h, 8, 16, HDIM);
    diff_prep_w<<<(8 * 2 * 512) / 256, 256, 0, stream>>>(Wo, wf + OFF_WOh, 8, 2, ADIM);
    diff_prep_state<<<(BATCH * SDIM) / 256, 256, 0, stream>>>(state, sf16);
    diff_sched<<<1, 32, 0, stream>>>(sched);
    diff_temb<<<TSTEPS, 256, 0, stream>>>(Wt1, bt1, Wt2, bt2, W1, b1, effb1);
    diff_init_x<<<(BATCH * ADIM) / 256, 256, 0, stream>>>(x);

    for (int s = TSTEPS - 1; s >= 0; --s)
        diff_step<<<BATCH / 64, 64, 0, stream>>>(wf, sf16, effb1, b2, b3, bo,
                                                 sched, x, out, s);
}
